// Encoder_89893665506098
// MI455X (gfx1250) — matrix operand order repaired and checked by reading, not compiled, not run on hardware
//
#include <hip/hip_runtime.h>

// ---------------------------------------------------------------------------
// Encoder on MI455X (gfx1250, wave32):
//   feats,W -> bf16 once;  Xe = feats @ W_embed^T + b_embed          (bf16 out)
//   G = Xe @ W_ih^T + b_ih + b_hh                                    (fp32, hoisted)
//   for t: gates = G[:,t,:] + h_bf16 @ W_hh^T ; LSTM cell fp32 ; out
// Matrix math: v_wmma_f32_16x16x32_bf16. Big GEMMs stage B tiles in LDS via
// GLOBAL_LOAD_ASYNC_TO_LDS_B128 (ASYNCcnt), fragment-shuffled for conflict-free
// ds_load_b128 reads.
// ---------------------------------------------------------------------------

typedef __bf16 bf16;
typedef __attribute__((ext_vector_type(16))) __bf16 v16bf;
typedef __attribute__((ext_vector_type(8)))  float  v8f;

#define Bb 256
#define Tt 64
#define Ff 2048
#define Ee 512
#define Hh 512

// --- bf16 WMMA fragment loaders (contiguous k = 16*(lane>>4)+idx order, same for A and B) -------
// A (16x32 MxK): lane m=lane&15, kb=16*(lane>>4);
//   elem idx = A[m, kb+idx] -> one contiguous 32B run -> 2x b128.
static __device__ __forceinline__ v16bf load_a_bf16(const bf16* __restrict__ A, int lda,
                                                    int row0, int k0, int lane) {
    const int m  = lane & 15;
    const int kb = (lane >> 4) * 16;
    const bf16* p = A + (size_t)(row0 + m) * lda + (k0 + kb);
    union { v16bf v; uint4 q[2]; } u;
    u.q[0] = *(const uint4*)p;
    u.q[1] = *(const uint4*)(p + 8);
    return u.v;
}
// B (32x16 KxN) from row-major weight W[N,K]: lane n=lane&15, ko=16*(lane>>4);
//   elem idx = W[n0+n, k0+ko+idx] -> one contiguous 32B run -> two b128.
static __device__ __forceinline__ v16bf load_b_bf16(const bf16* __restrict__ W, int ldb,
                                                    int n0, int k0, int lane) {
    const int n  = lane & 15;
    const int ko = (lane >> 4) * 16;
    const bf16* p = W + (size_t)(n0 + n) * ldb + (k0 + ko);
    union { v16bf v; uint4 q[2]; } u;
    u.q[0] = *(const uint4*)p;
    u.q[1] = *(const uint4*)(p + 8);
    return u.v;
}

static __device__ __forceinline__ v8f wmma_bf16(v16bf a, v16bf b, v8f c) {
    return __builtin_amdgcn_wmma_f32_16x16x32_bf16(false, a, false, b, (short)0, c, false, false);
}

static __device__ __forceinline__ float sigmoidf_(float x) {
    return 1.0f / (1.0f + __expf(-x));
}

// ---------------------------------------------------------------------------
// f32 -> bf16 bulk convert (vectorized x4; all sizes here are multiples of 4)
// ---------------------------------------------------------------------------
__global__ __launch_bounds__(256) void cvt_f32_to_bf16(const float* __restrict__ src,
                                                       bf16* __restrict__ dst, int n) {
    int i = (blockIdx.x * blockDim.x + threadIdx.x) * 4;
    if (i + 3 < n) {
        float4 f = *(const float4*)(src + i);
        union { bf16 h[4]; uint2 u; } p;
        p.h[0] = (bf16)f.x; p.h[1] = (bf16)f.y; p.h[2] = (bf16)f.z; p.h[3] = (bf16)f.w;
        *(uint2*)(dst + i) = p.u;
    } else {
        for (; i < n; ++i) dst[i] = (bf16)src[i];
    }
}

__global__ void zero_kernel(float* __restrict__ p, int n) {
    int i = blockIdx.x * blockDim.x + threadIdx.x;
    if (i < n) p[i] = 0.0f;
}

// ---------------------------------------------------------------------------
// C[M,N] = A[M,K](bf16) @ Bw[N,K](bf16)^T + bias1 (+bias2); C fp32 or bf16.
// Block tile 128x64, 256 threads = 8 waves (4 M-waves x 2 N-waves); wave 32x32.
// B tile (64 cols x 32 K = 4KB) double-buffered in LDS via async DMA:
//   wave w fetches subtile s=w>>1, half=w&1; lane l moves 16B
//   global  W[col0+s*16+(l&15), k0+16*(l>>4)+half*8 ..]  ->
//   LDS chunk[(s*2+half)*32 + l]  (fragment-shuffled: reads are lane*16B).
// A operand stays in registers (streamed global b128, software-pipelined).
// ---------------------------------------------------------------------------
template <bool OUT_BF16>
__global__ __launch_bounds__(256) void gemm_bf16_nt(
    const bf16* __restrict__ A, const bf16* __restrict__ Bw,
    const float* __restrict__ bias1, const float* __restrict__ bias2,
    void* __restrict__ Cout, int K, int lda, int ldb, int ldc)
{
    __shared__ bf16 sB[2 * 64 * 32];            // 2 buffers x 4KB

    const int lane = threadIdx.x & 31;
    const int wave = threadIdx.x >> 5;
    const int wm = wave & 3;
    const int wn = wave >> 2;
    const int colBase = blockIdx.x * 64;
    const int row0 = blockIdx.y * 128 + wm * 32;
    const int col0 = colBase + wn * 32;

    // ---- async B-stage issue: one b128 per lane per wave per K-step ----
    const int sub  = wave >> 1;                 // 0..3 : 16-col subtile
    const int half = wave & 1;                  // 0/1  : elems 0..7 / 8..15
    const unsigned sbase = (unsigned)(size_t)(&sB[0]);  // LDS byte offset (addr[31:0])
    const size_t growB = (size_t)(colBase + sub * 16 + (lane & 15)) * ldb
                       + (lane >> 4) * 16 + half * 8;
    const unsigned ldsoff = sbase + (unsigned)(((sub * 2 + half) * 32 + lane) * 16);

    #define ISSUE_B(buf, k0)                                                        \
        {                                                                           \
            unsigned long long ga = (unsigned long long)(size_t)(Bw + growB + (k0));\
            unsigned la = ldsoff + (unsigned)(buf) * 4096u;                         \
            asm volatile("global_load_async_to_lds_b128 %0, %1, off"                \
                         :: "v"(la), "v"(ga) : "memory");                           \
        }

    v8f acc00 = {}, acc01 = {}, acc10 = {}, acc11 = {};

    const int nsteps = K >> 5;                  // K is a multiple of 32, >= 512
    ISSUE_B(0, 0)
    ISSUE_B(1, 32)

    v16bf a0 = load_a_bf16(A, lda, row0,      0, lane);
    v16bf a1 = load_a_bf16(A, lda, row0 + 16, 0, lane);

    const int m  = lane & 15;
    const int kb = (lane >> 4) * 8;
    const uint4* sBq = (const uint4*)sB;        // chunk view (16B units)

    for (int i = 0; i < nsteps; ++i) {
        const int k0 = i << 5;
        // wait for this buffer's DMA (own wave), then align whole workgroup
        asm volatile("s_wait_asynccnt 0x1" ::: "memory");
        __syncthreads();

        const int cb = (i & 1) * 256;           // 256 chunks per buffer
        union { v16bf v; uint4 q[2]; } ub0, ub1;
        ub0.q[0] = sBq[cb + (wn * 2 + 0) * 64 +  0 + lane];   // sub=2*wn, half 0
        ub0.q[1] = sBq[cb + (wn * 2 + 0) * 64 + 32 + lane];   //          half 1
        ub1.q[0] = sBq[cb + (wn * 2 + 1) * 64 +  0 + lane];   // sub=2*wn+1
        ub1.q[1] = sBq[cb + (wn * 2 + 1) * 64 + 32 + lane];

        // pipeline next A fragments + streaming prefetch (branchless clamp)
        int kn = k0 + 32; kn = (kn < K) ? kn : k0;
        int kp = k0 + 256; kp = (kp < K) ? kp : k0;
        __builtin_prefetch(A + (size_t)(row0 + m) * lda + (kp + kb), 0, 1);
        __builtin_prefetch(A + (size_t)(row0 + 16 + m) * lda + (kp + kb), 0, 1);
        v16bf na0 = load_a_bf16(A, lda, row0,      kn, lane);
        v16bf na1 = load_a_bf16(A, lda, row0 + 16, kn, lane);

        acc00 = wmma_bf16(a0, ub0.v, acc00);
        acc01 = wmma_bf16(a0, ub1.v, acc01);
        acc10 = wmma_bf16(a1, ub0.v, acc10);
        acc11 = wmma_bf16(a1, ub1.v, acc11);

        __syncthreads();                        // all waves done reading buffer
        if (i + 2 < nsteps) ISSUE_B(i & 1, (i + 2) << 5)

        a0 = na0; a1 = na1;
    }
    #undef ISSUE_B

    const int lh = lane >> 4;                   // C: M = r + 8*lh, N = lane&15
    const int ln = lane & 15;
    #pragma unroll
    for (int sn = 0; sn < 2; ++sn) {
        const int col = col0 + sn * 16 + ln;
        const float bsum = bias1[col] + (bias2 ? bias2[col] : 0.0f);
        #pragma unroll
        for (int sm = 0; sm < 2; ++sm) {
            v8f acc = (sm == 0) ? (sn == 0 ? acc00 : acc01)
                                : (sn == 0 ? acc10 : acc11);
            #pragma unroll
            for (int r = 0; r < 8; ++r) {
                const int row = row0 + sm * 16 + r + lh * 8;
                const float v = acc[r] + bsum;
                if (OUT_BF16) ((bf16*)Cout)[(size_t)row * ldc + col] = (bf16)v;
                else          ((float*)Cout)[(size_t)row * ldc + col] = v;
            }
        }
    }
}

// ---------------------------------------------------------------------------
// One LSTM step: gates[b, g*H+n] = G[b*T+t, g*H+n] + sum_k h[b,k]*W_hh[g*H+n,k]
// Block 64x32, 8 waves; wave = one 16x16 tile x 4 gate accumulators.
// Grid (16,4) = 64 WGs. W_hh (2MB bf16) and h (256KB) are L2/L0 resident, so
// direct global fragment loads (no barriers on the sequential critical path).
// ---------------------------------------------------------------------------
__global__ __launch_bounds__(256) void lstm_step_kernel(
    const float* __restrict__ G, const bf16* __restrict__ Whh,
    const bf16* __restrict__ h_in, bf16* __restrict__ h_out,
    float* __restrict__ c_state, float* __restrict__ out, int t)
{
    const int lane = threadIdx.x & 31;
    const int wave = threadIdx.x >> 5;
    const int wm = wave & 3;
    const int wn = wave >> 2;
    const int b0 = blockIdx.y * 64 + wm * 16;
    const int n0 = blockIdx.x * 32 + wn * 16;

    v8f acc_i = {}, acc_f = {}, acc_g = {}, acc_o = {};

    v16bf a   = load_a_bf16(h_in, Hh, b0, 0, lane);
    v16bf bi  = load_b_bf16(Whh, Hh, 0 * Hh + n0, 0, lane);
    v16bf bf_ = load_b_bf16(Whh, Hh, 1 * Hh + n0, 0, lane);
    v16bf bg  = load_b_bf16(Whh, Hh, 2 * Hh + n0, 0, lane);
    v16bf bo  = load_b_bf16(Whh, Hh, 3 * Hh + n0, 0, lane);

    for (int k0 = 0; k0 < Hh; k0 += 32) {
        int kn = k0 + 32;
        kn = (kn < Hh) ? kn : k0;
        v16bf na  = load_a_bf16(h_in, Hh, b0, kn, lane);
        v16bf nbi = load_b_bf16(Whh, Hh, 0 * Hh + n0, kn, lane);
        v16bf nbf = load_b_bf16(Whh, Hh, 1 * Hh + n0, kn, lane);
        v16bf nbg = load_b_bf16(Whh, Hh, 2 * Hh + n0, kn, lane);
        v16bf nbo = load_b_bf16(Whh, Hh, 3 * Hh + n0, kn, lane);

        acc_i = wmma_bf16(a, bi,  acc_i);
        acc_f = wmma_bf16(a, bf_, acc_f);
        acc_g = wmma_bf16(a, bg,  acc_g);
        acc_o = wmma_bf16(a, bo,  acc_o);

        a = na; bi = nbi; bf_ = nbf; bg = nbg; bo = nbo;
    }

    const int lh = lane >> 4;
    const int n  = n0 + (lane & 15);
    #pragma unroll
    for (int r = 0; r < 8; ++r) {
        const int b = b0 + r + lh * 8;
        const size_t grow = ((size_t)b * Tt + t) * (4 * Hh);
        const float gi = acc_i[r] + G[grow + 0 * Hh + n];
        const float gf = acc_f[r] + G[grow + 1 * Hh + n];
        const float gg = acc_g[r] + G[grow + 2 * Hh + n];
        const float go = acc_o[r] + G[grow + 3 * Hh + n];
        const float ig = sigmoidf_(gi);
        const float fg = sigmoidf_(gf);
        const float g  = tanhf(gg);
        const float og = sigmoidf_(go);
        const size_t hidx = (size_t)b * Hh + n;
        const float cn = fg * c_state[hidx] + ig * g;
        c_state[hidx] = cn;
        const float hn = og * tanhf(cn);
        h_out[hidx] = (bf16)hn;                         // next step's A operand
        out[((size_t)b * Tt + t) * Hh + n] = hn;        // out[b, t, n] fp32
    }
}

// ---------------------------------------------------------------------------
extern "C" void kernel_launch(void* const* d_in, const int* in_sizes, int n_in,
                              void* d_out, int out_size, void* d_ws, size_t ws_size,
                              hipStream_t stream) {
    const float* feats   = (const float*)d_in[0];  // [16384, 2048]
    const float* W_embed = (const float*)d_in[1];  // [512, 2048]
    const float* b_embed = (const float*)d_in[2];
    const float* W_ih    = (const float*)d_in[3];  // [2048, 512]
    const float* W_hh    = (const float*)d_in[4];  // [2048, 512]
    const float* b_ih    = (const float*)d_in[5];
    const float* b_hh    = (const float*)d_in[6];
    float* out = (float*)d_out;                    // [B,T,H] fp32

    // workspace layout (256B-aligned chunks)
    char* ws = (char*)d_ws;
    size_t off = 0;
    auto take = [&](size_t bytes) { char* p = ws + off; off += (bytes + 255) & ~(size_t)255; return p; };
    bf16*  featsbf = (bf16*)take((size_t)Bb * Tt * Ff * 2);      // 67.1 MB
    bf16*  Webf    = (bf16*)take((size_t)Ee * Ff * 2);           // 2 MB
    bf16*  Wibf    = (bf16*)take((size_t)4 * Hh * Ee * 2);       // 2 MB
    bf16*  Whbf    = (bf16*)take((size_t)4 * Hh * Hh * 2);       // 2 MB
    bf16*  Xe      = (bf16*)take((size_t)Bb * Tt * Ee * 2);      // 16.8 MB
    float* G       = (float*)take((size_t)Bb * Tt * 4 * Hh * 4); // 134.2 MB
    bf16*  hA      = (bf16*)take((size_t)Bb * Hh * 2);
    bf16*  hB      = (bf16*)take((size_t)Bb * Hh * 2);
    float* cS      = (float*)take((size_t)Bb * Hh * 4);

    // one-time precision conversion (weights then stay L2-resident)
    {
        int n = Bb * Tt * Ff;
        cvt_f32_to_bf16<<<(n / 4 + 255) / 256, 256, 0, stream>>>(feats, featsbf, n);
        n = Ee * Ff;
        cvt_f32_to_bf16<<<(n / 4 + 255) / 256, 256, 0, stream>>>(W_embed, Webf, n);
        n = 4 * Hh * Ee;
        cvt_f32_to_bf16<<<(n / 4 + 255) / 256, 256, 0, stream>>>(W_ih, Wibf, n);
        n = 4 * Hh * Hh;
        cvt_f32_to_bf16<<<(n / 4 + 255) / 256, 256, 0, stream>>>(W_hh, Whbf, n);
    }
    // zero h0 (bf16, both buffers) and c (fp32): contiguous region
    {
        const int nfloat = (2 * Bb * Hh * 2 + Bb * Hh * 4) / 4;
        zero_kernel<<<(nfloat + 255) / 256, 256, 0, stream>>>((float*)hA, nfloat);
    }

    // Xe(bf16) = feats @ W_embed^T + b_embed
    gemm_bf16_nt<true><<<dim3(Ee / 64, (Bb * Tt) / 128), 256, 0, stream>>>(
        featsbf, Webf, b_embed, nullptr, Xe, Ff, Ff, Ff, Ee);

    // G(fp32) = Xe @ W_ih^T + b_ih + b_hh
    gemm_bf16_nt<false><<<dim3((4 * Hh) / 64, (Bb * Tt) / 128), 256, 0, stream>>>(
        Xe, Wibf, b_ih, b_hh, G, Ee, Ee, Ee, 4 * Hh);

    // sequential LSTM steps (h double-buffered bf16; c fp32 in place)
    bf16* hin = hA;
    bf16* hout = hB;
    for (int t = 0; t < Tt; ++t) {
        lstm_step_kernel<<<dim3(Hh / 32, Bb / 64), 256, 0, stream>>>(
            G, Whbf, hin, hout, cS, out, t);
        bf16* tmp = hin; hin = hout; hout = tmp;
    }
}
